// MixtureOfExperts_19035295056349
// MI455X (gfx1250) — matrix operand order repaired and checked by reading, not compiled, not run on hardware
//
#include <hip/hip_runtime.h>

#define NT   8192      // B*T tokens
#define C    1024
#define F    4096
#define E    8
#define CAP  8192
#define TILE_M 32
#define FH   2048      // F half processed per pass

typedef __attribute__((ext_vector_type(16))) __bf16 v16bf;
typedef __attribute__((ext_vector_type(8)))  __bf16 v8bf;
typedef __attribute__((ext_vector_type(4)))  __bf16 v4bf;
typedef __attribute__((ext_vector_type(8)))  float  v8f;
typedef __attribute__((ext_vector_type(4)))  float  v4f;

static __device__ __forceinline__ v8f wmma_bf16(v16bf a, v16bf b, v8f c) {
    return __builtin_amdgcn_wmma_f32_16x16x32_bf16(false, a, false, b, (short)0, c, false, false);
}
static __device__ __forceinline__ v16bf mk16(v8bf lo, v8bf hi) {
    v16bf r;
#pragma unroll
    for (int j = 0; j < 8; ++j) { r[j] = lo[j]; r[8 + j] = hi[j]; }
    return r;
}

// ---------------------------------------------------------------- zero init
__global__ void moe_zero(float* __restrict__ out, int* __restrict__ cnt, long n) {
    long i = (long)blockIdx.x * blockDim.x + threadIdx.x;
    if (i < n) out[i] = 0.0f;
    if (i < E) cnt[i] = 0;
}

// ---------------------------------------------------------------- weight shuffle
// fp32 [E][K][N] row-major  ->  bf16 WMMA-B fragment blocks:
// block (e,ks,nt) = 1KB: lane L holds 16 bf16 = column nt*16+(L&15),
// K = ks*32 + (L>>4)*16 + j  (contiguous k order, same as the A tile reads). One wave per block.
__global__ void moe_shuffle_w(const float* __restrict__ src, __bf16* __restrict__ dst,
                              int K, int N) {
    int wid  = (blockIdx.x * blockDim.x + threadIdx.x) >> 5;
    int lane = threadIdx.x & 31;
    int blocksPerE = (K >> 5) * (N >> 4);
    int e    = wid / blocksPerE;
    int rem  = wid - e * blocksPerE;
    int ks   = rem / (N >> 4);
    int nt   = rem - ks * (N >> 4);
    int k0   = ks * 32 + ((lane >> 4) << 4);
    int n    = nt * 16 + (lane & 15);
    const float* sp = src + ((size_t)e * K + k0) * N + n;
    float v[16];
#pragma unroll
    for (int j = 0; j < 16; ++j) v[j] = sp[(size_t)j * N];
    v8bf lo, hi;
#pragma unroll
    for (int j = 0; j < 8; ++j) { lo[j] = (__bf16)v[j]; hi[j] = (__bf16)v[8 + j]; }
    __bf16* dp = dst + (size_t)wid * 512 + lane * 16;
    *(v8bf*)dp       = lo;
    *(v8bf*)(dp + 8) = hi;
}

// ---------------------------------------------------------------- router
__global__ void moe_router(const float* __restrict__ x,
                           const float* __restrict__ gw,
                           int* __restrict__ cnt,
                           int* __restrict__ lists,
                           float* __restrict__ wts) {
    int wave = threadIdx.x >> 5;
    int lane = threadIdx.x & 31;
    int tok  = blockIdx.x * (blockDim.x >> 5) + wave;
    if (tok >= NT) return;
    float acc[E];
#pragma unroll
    for (int e = 0; e < E; ++e) acc[e] = 0.f;
    const float* xp = x + (size_t)tok * C;
    for (int c = lane; c < C; c += 32) {
        float xv = xp[c];
        const float* g = gw + (size_t)c * E;
#pragma unroll
        for (int e = 0; e < E; ++e) acc[e] += xv * g[e];
    }
#pragma unroll
    for (int e = 0; e < E; ++e) {
#pragma unroll
        for (int off = 16; off > 0; off >>= 1)
            acc[e] += __shfl_xor(acc[e], off, 32);
    }
    if (lane == 0) {
        float m = acc[0];
#pragma unroll
        for (int e = 1; e < E; ++e) m = fmaxf(m, acc[e]);
        float p[E];
#pragma unroll
        for (int e = 0; e < E; ++e) p[e] = __expf(acc[e] - m);
        int i1 = 0;
#pragma unroll
        for (int e = 1; e < E; ++e) if (p[e] > p[i1]) i1 = e;
        int i2 = (i1 == 0) ? 1 : 0;
#pragma unroll
        for (int e = 0; e < E; ++e) if (e != i1 && p[e] > p[i2]) i2 = e;
        float ps = p[i1] + p[i2];
        int pos1 = atomicAdd(&cnt[i1], 1);
        lists[i1 * CAP + pos1] = tok;  wts[i1 * CAP + pos1] = p[i1] / ps;
        int pos2 = atomicAdd(&cnt[i2], 1);
        lists[i2 * CAP + pos2] = tok;  wts[i2 * CAP + pos2] = p[i2] / ps;
    }
}

// ---------------------------------------------------------------- expert FFN
#define XS_STRIDE 1032   // bf16 elems (2064B row, 16B aligned, bank-spread)
#define HS_STRIDE 2056   // bf16 elems (4112B row, 16B aligned, bank-spread)
#define XS_OFF    0
#define HS_OFF    (TILE_M * XS_STRIDE * 2)                  // 66048
#define TOK_OFF   (HS_OFF + TILE_M * HS_STRIDE * 2)         // 197632
#define WT_OFF    (TOK_OFF + TILE_M * 4)
#define LDS_BYTES (WT_OFF + TILE_M * 4)                     // 197888

__global__ void __launch_bounds__(256, 1)
moe_expert(const float*  __restrict__ x,
           const __bf16* __restrict__ w1s,
           const __bf16* __restrict__ w2s,
           const int*    __restrict__ cnt,
           const int*    __restrict__ lists,
           const float*  __restrict__ wts,
           float* __restrict__ out) {
    extern __shared__ char smem[];
    __bf16* Xs    = (__bf16*)(smem + XS_OFF);
    __bf16* Hs    = (__bf16*)(smem + HS_OFF);
    int*    tok_s = (int*)  (smem + TOK_OFF);
    float*  wt_s  = (float*)(smem + WT_OFF);

    const int e    = blockIdx.y;
    const int n_e  = cnt[e];
    const int row0 = blockIdx.x * TILE_M;
    if (row0 >= n_e) return;

    const int tid  = threadIdx.x;
    const int wave = tid >> 5;
    const int lane = tid & 31;
    const int lhi  = lane >> 4;
    const int llo  = lane & 15;

    if (tid < TILE_M) {
        int row = row0 + tid;
        if (row < n_e) { tok_s[tid] = lists[e * CAP + row]; wt_s[tid] = wts[e * CAP + row]; }
        else           { tok_s[tid] = 0;                    wt_s[tid] = 0.f; }
    }
    __syncthreads();

    // gather X tile -> bf16 LDS, 4 cols per thread (float4 load, 8B bf16 store)
    for (int i = tid * 4; i < TILE_M * C; i += 256 * 4) {
        int r = i >> 10;
        int c = i & (C - 1);
        v4f v = {0.f, 0.f, 0.f, 0.f};
        if (row0 + r < n_e) v = *(const v4f*)(x + (size_t)tok_s[r] * C + c);
        v4bf b;
#pragma unroll
        for (int j = 0; j < 4; ++j) b[j] = (__bf16)v[j];
        *(v4bf*)(Xs + r * XS_STRIDE + c) = b;
    }
    __syncthreads();

    v8f accO[16];            // persistent out accumulators: [nb(2)][t(4)][mt(2)]
    const v8f vzero = {0.f,0.f,0.f,0.f,0.f,0.f,0.f,0.f};
#pragma unroll
    for (int i = 0; i < 16; ++i) accO[i] = vzero;

    const size_t w1e_blk = ((size_t)e * (C / 32)) * (F / 16);   // block index base
    const size_t w2e_blk = ((size_t)e * (F / 32)) * (C / 16);

    for (int half = 0; half < 2; ++half) {
        // -------- Phase 1: Hs[32, FH] = gelu(X @ W1e[:, half*FH .. +FH))
        for (int nb = 0; nb < 4; ++nb) {                 // wave owns 256 cols of half
            const int nloc  = wave * 256 + nb * 64;      // col within half
            const int ntile = (half * FH + nloc) >> 4;   // global 16-col tile idx
            v8f acc[8];                                  // [t][mt]
#pragma unroll
            for (int i = 0; i < 8; ++i) acc[i] = vzero;
            for (int k = 0; k < C; k += 32) {
                const __bf16* ap = Xs + llo * XS_STRIDE + k + lhi * 16;
                v16bf a0 = mk16(*(const v8bf*)ap, *(const v8bf*)(ap + 8));
                const __bf16* aq = ap + 16 * XS_STRIDE;
                v16bf a1 = mk16(*(const v8bf*)aq, *(const v8bf*)(aq + 8));
                const __bf16* bbase =
                    w1s + ((w1e_blk + (size_t)(k >> 5) * (F / 16)) + ntile) * 512 + lane * 16;
#pragma unroll
                for (int t = 0; t < 4; ++t) {
                    const __bf16* bp = bbase + (size_t)t * 512;
                    v16bf b = mk16(*(const v8bf*)bp, *(const v8bf*)(bp + 8));
                    acc[t * 2 + 0] = wmma_bf16(a0, b, acc[t * 2 + 0]);
                    acc[t * 2 + 1] = wmma_bf16(a1, b, acc[t * 2 + 1]);
                }
            }
            // exact GELU -> bf16 Hs
#pragma unroll
            for (int t = 0; t < 4; ++t)
#pragma unroll
                for (int mt = 0; mt < 2; ++mt)
#pragma unroll
                    for (int r = 0; r < 8; ++r) {
                        float v = acc[t * 2 + mt][r];
                        float g = 0.5f * v * (1.f + erff(v * 0.70710678118654752f));
                        int m = mt * 16 + r + (lhi << 3);
                        int n = nloc + t * 16 + llo;
                        Hs[m * HS_STRIDE + n] = (__bf16)g;
                    }
        }
        __syncthreads();

        // -------- Phase 2: accO += Hs @ W2e[half*FH .. +FH, :]
#pragma unroll
        for (int nb = 0; nb < 2; ++nb) {                 // wave owns 128 out cols
            const int ntile = ((wave * 128 + nb * 64) >> 4);
            for (int k = 0; k < FH; k += 32) {
                const __bf16* ap = Hs + llo * HS_STRIDE + k + lhi * 16;
                v16bf a0 = mk16(*(const v8bf*)ap, *(const v8bf*)(ap + 8));
                const __bf16* aq = ap + 16 * HS_STRIDE;
                v16bf a1 = mk16(*(const v8bf*)aq, *(const v8bf*)(aq + 8));
                const int kg = half * FH + k;
                const __bf16* bbase =
                    w2s + ((w2e_blk + (size_t)(kg >> 5) * (C / 16)) + ntile) * 512 + lane * 16;
#pragma unroll
                for (int t = 0; t < 4; ++t) {
                    const __bf16* bp = bbase + (size_t)t * 512;
                    v16bf b = mk16(*(const v8bf*)bp, *(const v8bf*)(bp + 8));
                    accO[(nb * 4 + t) * 2 + 0] = wmma_bf16(a0, b, accO[(nb * 4 + t) * 2 + 0]);
                    accO[(nb * 4 + t) * 2 + 1] = wmma_bf16(a1, b, accO[(nb * 4 + t) * 2 + 1]);
                }
            }
        }
        __syncthreads();   // Hs reused next half
    }

    // -------- weighted scatter-add
#pragma unroll
    for (int nb = 0; nb < 2; ++nb)
#pragma unroll
        for (int t = 0; t < 4; ++t)
#pragma unroll
            for (int mt = 0; mt < 2; ++mt)
#pragma unroll
                for (int r = 0; r < 8; ++r) {
                    int m = mt * 16 + r + (lhi << 3);
                    int n = wave * 128 + nb * 64 + t * 16 + llo;
                    float v = accO[(nb * 4 + t) * 2 + mt][r] * wt_s[m];
                    atomicAdd(&out[(size_t)tok_s[m] * C + n], v);
                }
}

// ---------------------------------------------------------------- launch
extern "C" void kernel_launch(void* const* d_in, const int* in_sizes, int n_in,
                              void* d_out, int out_size, void* d_ws, size_t ws_size,
                              hipStream_t stream) {
    const float* x  = (const float*)d_in[0];
    const float* gw = (const float*)d_in[1];
    const float* w1 = (const float*)d_in[2];
    const float* w2 = (const float*)d_in[3];
    float* out = (float*)d_out;

    int*    cnt   = (int*)d_ws;
    int*    lists = (int*)((char*)d_ws + 128);
    float*  wts   = (float*)((char*)d_ws + 128 + (size_t)E * CAP * sizeof(int));
    __bf16* w1s   = (__bf16*)((char*)d_ws + (1u << 20));
    __bf16* w2s   = (__bf16*)((char*)d_ws + (1u << 20) + (size_t)E * C * F * 2);

    (void)in_sizes; (void)n_in; (void)out_size; (void)ws_size;

    static bool attr_set = false;
    if (!attr_set) {
        hipFuncSetAttribute((const void*)moe_expert,
                            hipFuncAttributeMaxDynamicSharedMemorySize, LDS_BYTES);
        attr_set = true;
    }

    long n = (long)NT * C;
    moe_zero<<<(n + 255) / 256, 256, 0, stream>>>(out, cnt, n);

    // bf16 fragment-shuffle of weights (E*(K/32)*(N/16) waves each)
    {
        int waves1 = E * (C / 32) * (F / 16);
        moe_shuffle_w<<<waves1 * 32 / 256, 256, 0, stream>>>(w1, w1s, C, F);
        int waves2 = E * (F / 32) * (C / 16);
        moe_shuffle_w<<<waves2 * 32 / 256, 256, 0, stream>>>(w2, w2s, F, C);
    }

    moe_router<<<NT / 8, 256, 0, stream>>>(x, gw, cnt, lists, wts);

    dim3 grid(CAP / TILE_M, E);
    moe_expert<<<grid, 256, LDS_BYTES, stream>>>(x, w1s, w2s, cnt, lists, wts, out);
}
